// SwinLikeBlock_18245021073657
// MI455X (gfx1250) — matrix operand order repaired and checked by reading, not compiled, not run on hardware
//
#include <hip/hip_runtime.h>
#include <hip/hip_bf16.h>

// ---------------------------------------------------------------------------
// Swin-like block on gfx1250 (MI455X): f16 WMMA GEMMs + wave-per-head window
// attention. D = A*B + C via v_wmma_f32_16x16x32_f16, wave32.
// ---------------------------------------------------------------------------

typedef __attribute__((ext_vector_type(16))) _Float16 v16h;
typedef __attribute__((ext_vector_type(8)))  _Float16 v8h;
typedef __attribute__((ext_vector_type(8)))  float    v8f;

#define N_TOK   100352      // 8 * 112 * 112
#define CCH     256
#define NQKV    768
#define HID     1024
#define HEADS   8
#define DH      32
#define WSZ     7
#define LW      49          // tokens per window
#define IMG_L   12544       // 112*112
#define IMG_W   112
#define NWIN_S  16          // windows per side
#define EPS_LN  1e-5f

__device__ __forceinline__ v16h zero16() {
    v16h z;
#pragma unroll
    for (int i = 0; i < 16; ++i) z[i] = (_Float16)0.0f;
    return z;
}
__device__ __forceinline__ v8f zero8f() {
    v8f z;
#pragma unroll
    for (int i = 0; i < 8; ++i) z[i] = 0.0f;
    return z;
}

// A-fragment (16x32 f16, MxK): lane group g = lane/16.
// elements 0..15 = K[16g .. 16g+15]
__device__ __forceinline__ v16h load_a_frag(const _Float16* rowk0, int g) {
    v8h lo = *(const v8h*)(rowk0 + 16 * g);
    v8h hi = *(const v8h*)(rowk0 + 16 * g + 8);
    v16h r;
#pragma unroll
    for (int i = 0; i < 8; ++i) { r[i] = lo[i]; r[i + 8] = hi[i]; }
    return r;
}

__device__ __forceinline__ v8f wmma_f16(v16h a, v16h b, v8f c) {
    return __builtin_amdgcn_wmma_f32_16x16x32_f16(false, a, false, b,
                                                  (short)0, c, false, false);
}

// ---------------------------------------------------------------------------
// f32 -> f16 weight conversion
// ---------------------------------------------------------------------------
__global__ void cvt_f32_f16_kernel(const float* __restrict__ in,
                                   _Float16* __restrict__ out, int n) {
    int i = blockIdx.x * blockDim.x + threadIdx.x;
    int stride = gridDim.x * blockDim.x;
    for (; i < n; i += stride) out[i] = (_Float16)in[i];
}

// ---------------------------------------------------------------------------
// LayerNorm over C=256, one token per 256-thread block, f16 output
// ---------------------------------------------------------------------------
__global__ void ln_kernel(const float* __restrict__ x,
                          const float* __restrict__ g,
                          const float* __restrict__ b,
                          _Float16* __restrict__ out) {
    __shared__ float red[256];
    __shared__ float stats[2];
    const int t = blockIdx.x;
    const int c = threadIdx.x;
    const float v = x[(size_t)t * CCH + c];

    red[c] = v;
    __syncthreads();
#pragma unroll
    for (int s = 128; s > 0; s >>= 1) {
        if (c < s) red[c] += red[c + s];
        __syncthreads();
    }
    if (c == 0) stats[0] = red[0] * (1.0f / CCH);
    __syncthreads();
    const float mean = stats[0];
    const float d = v - mean;

    red[c] = d * d;
    __syncthreads();
#pragma unroll
    for (int s = 128; s > 0; s >>= 1) {
        if (c < s) red[c] += red[c + s];
        __syncthreads();
    }
    if (c == 0) stats[1] = __frsqrt_rn(red[0] * (1.0f / CCH) + EPS_LN);
    __syncthreads();

    out[(size_t)t * CCH + c] = (_Float16)(d * stats[1] * g[c] + b[c]);
}

// ---------------------------------------------------------------------------
// Wave-tiled GEMM: C[M,N] = A[M,K] * W[N,K]^T (+ epilogue)
//   MODE 0: out f16 = acc + bias                      (qkv)
//   MODE 1: out f32 = resid + acc + bias              (proj + residual)
//   MODE 2: out f16 = gelu_erf(acc + bias)            (fc1)
//   MODE 3: out f32 += acc + bias                     (fc2 + residual)
// One wave computes a 32x32 tile (2x2 WMMA accumulators). M,N,K % 32 == 0.
// ---------------------------------------------------------------------------
template <int MODE>
__global__ void gemm_wmma_kernel(const _Float16* __restrict__ A,
                                 const _Float16* __restrict__ W,
                                 const float* __restrict__ bias,
                                 const float* __restrict__ resid,
                                 void* __restrict__ out,
                                 int N, int K) {
    const int wid   = (blockIdx.x * blockDim.x + threadIdx.x) >> 5;
    const int lane  = threadIdx.x & 31;
    const int g     = lane >> 4;
    const int lm    = lane & 15;
    const int nt32  = N >> 5;
    const int m0    = (wid / nt32) << 5;
    const int n0    = (wid % nt32) << 5;

    const _Float16* arow0 = A + (size_t)(m0 + lm) * K;
    const _Float16* arow1 = A + (size_t)(m0 + 16 + lm) * K;
    const _Float16* wrow0 = W + (size_t)(n0 + lm) * K;
    const _Float16* wrow1 = W + (size_t)(n0 + 16 + lm) * K;

    v8f acc00 = zero8f(), acc01 = zero8f(), acc10 = zero8f(), acc11 = zero8f();

    for (int k0 = 0; k0 < K; k0 += 32) {
        __builtin_prefetch(arow0 + k0 + 128, 0, 1);
        __builtin_prefetch(arow1 + k0 + 128, 0, 1);
        v16h a0 = load_a_frag(arow0 + k0, g);
        v16h a1 = load_a_frag(arow1 + k0, g);
        v16h b0 = *(const v16h*)(wrow0 + k0 + 16 * g);
        v16h b1 = *(const v16h*)(wrow1 + k0 + 16 * g);
        acc00 = wmma_f16(a0, b0, acc00);
        acc01 = wmma_f16(a0, b1, acc01);
        acc10 = wmma_f16(a1, b0, acc10);
        acc11 = wmma_f16(a1, b1, acc11);
    }

    v8f accs[2][2] = {{acc00, acc01}, {acc10, acc11}};
#pragma unroll
    for (int mi = 0; mi < 2; ++mi) {
#pragma unroll
        for (int ni = 0; ni < 2; ++ni) {
            const int n = n0 + ni * 16 + lm;
            const float bn = bias[n];
#pragma unroll
            for (int r = 0; r < 8; ++r) {
                const int m = m0 + mi * 16 + r + 8 * g;
                const size_t idx = (size_t)m * N + n;
                float v = accs[mi][ni][r] + bn;
                if (MODE == 0) {
                    ((_Float16*)out)[idx] = (_Float16)v;
                } else if (MODE == 1) {
                    ((float*)out)[idx] = resid[idx] + v;
                } else if (MODE == 2) {
                    v = 0.5f * v * (1.0f + erff(v * 0.70710678118654752f));
                    ((_Float16*)out)[idx] = (_Float16)v;
                } else {
                    ((float*)out)[idx] += v;
                }
            }
        }
    }
}

// ---------------------------------------------------------------------------
// Window attention: one wave (32 threads) per (window, head).
// qkv rows: [q(256) | k(256) | v(256)], head h at offset h*32 in each.
// L=49 padded to 64; S via 16 WMMAs, softmax, O = P*V via 16 WMMAs.
// ---------------------------------------------------------------------------
__global__ void win_attn_kernel(const _Float16* __restrict__ qkv,
                                _Float16* __restrict__ outb) {
    __shared__ __attribute__((aligned(32))) float    S[64 * 64];
    __shared__ __attribute__((aligned(32))) _Float16 P[64 * 64];
    __shared__ __attribute__((aligned(32))) _Float16 Vt[32 * 64];

    const int w    = blockIdx.x >> 3;     // window id
    const int h    = blockIdx.x & 7;      // head id
    const int img  = w / (NWIN_S * NWIN_S);
    const int rem  = w % (NWIN_S * NWIN_S);
    const int wh   = rem / NWIN_S;
    const int wc   = rem % NWIN_S;
    const int base = img * IMG_L + wh * WSZ * IMG_W + wc * WSZ;

    const int lane = threadIdx.x;
    const int g    = lane >> 4;
    const int lm   = lane & 15;
    const float scale = 0.17677669529663689f;   // 32^-0.5

    // window-local token -> global token row
    auto tok = [&](int wl) -> int {
        return base + (wl / WSZ) * IMG_W + (wl % WSZ);
    };

    // ---- S = scale * Q K^T (64x64 padded), column mask -> -inf ----
    v16h bfr[4];
#pragma unroll
    for (int nt = 0; nt < 4; ++nt) {
        const int ncol = nt * 16 + lm;
        if (ncol < LW) {
            const _Float16* kr = qkv + (size_t)tok(ncol) * NQKV + CCH + h * DH;
            bfr[nt] = *(const v16h*)(kr + 16 * g);
        } else {
            bfr[nt] = zero16();
        }
    }
#pragma unroll
    for (int mt = 0; mt < 4; ++mt) {
        const int mrow = mt * 16 + lm;
        v16h a;
        if (mrow < LW) {
            a = load_a_frag(qkv + (size_t)tok(mrow) * NQKV + h * DH, g);
        } else {
            a = zero16();
        }
#pragma unroll
        for (int nt = 0; nt < 4; ++nt) {
            v8f c = zero8f();
            c = wmma_f16(a, bfr[nt], c);
            const int n = nt * 16 + lm;
#pragma unroll
            for (int r = 0; r < 8; ++r) {
                const int m = mt * 16 + r + 8 * g;
                float v = c[r] * scale;
                if (n >= LW) v = -1.0e30f;
                S[m * 64 + n] = v;
            }
        }
    }

    // ---- stage V^T (32 x 64), zero-padded tokens ----
#pragma unroll
    for (int ph = 0; ph < 2; ++ph) {
        const int t = ph * 32 + lane;
        if (t < LW) {
            const _Float16* vr = qkv + (size_t)tok(t) * NQKV + 2 * CCH + h * DH;
#pragma unroll
            for (int d = 0; d < DH; ++d) Vt[d * 64 + t] = vr[d];
        } else {
#pragma unroll
            for (int d = 0; d < DH; ++d) Vt[d * 64 + t] = (_Float16)0.0f;
        }
    }
    __syncthreads();

    // ---- softmax rows 0..48 (lane owns cols {lane, lane+32}) ----
    for (int m = 0; m < LW; ++m) {
        float x0 = S[m * 64 + lane];
        float x1 = S[m * 64 + 32 + lane];
        float mx = fmaxf(x0, x1);
#pragma unroll
        for (int off = 16; off >= 1; off >>= 1)
            mx = fmaxf(mx, __shfl_xor(mx, off, 32));
        float e0 = __expf(x0 - mx);
        float e1 = __expf(x1 - mx);
        float s = e0 + e1;
#pragma unroll
        for (int off = 16; off >= 1; off >>= 1)
            s += __shfl_xor(s, off, 32);
        const float inv = 1.0f / s;
        P[m * 64 + lane]      = (_Float16)(e0 * inv);
        P[m * 64 + 32 + lane] = (_Float16)(e1 * inv);
    }
    for (int m = LW; m < 64; ++m) {
        P[m * 64 + lane]      = (_Float16)0.0f;
        P[m * 64 + 32 + lane] = (_Float16)0.0f;
    }
    __syncthreads();

    // ---- O = P * V  (64x32, K=64 in two steps) ----
#pragma unroll
    for (int mt = 0; mt < 4; ++mt) {
#pragma unroll
        for (int nt = 0; nt < 2; ++nt) {
            v8f acc = zero8f();
#pragma unroll
            for (int ks = 0; ks < 2; ++ks) {
                v16h a = load_a_frag(&P[(mt * 16 + lm) * 64 + ks * 32], g);
                v16h b = *(const v16h*)(&Vt[(nt * 16 + lm) * 64 + ks * 32 + 16 * g]);
                acc = wmma_f16(a, b, acc);
            }
#pragma unroll
            for (int r = 0; r < 8; ++r) {
                const int m = mt * 16 + r + 8 * g;
                if (m < LW) {
                    const size_t o = (size_t)tok(m) * CCH + h * DH + nt * 16 + lm;
                    outb[o] = (_Float16)acc[r];
                }
            }
        }
    }
}

// ---------------------------------------------------------------------------
// Host-side launcher
// ---------------------------------------------------------------------------
extern "C" void kernel_launch(void* const* d_in, const int* in_sizes, int n_in,
                              void* d_out, int out_size, void* d_ws, size_t ws_size,
                              hipStream_t stream) {
    (void)in_sizes; (void)n_in; (void)out_size; (void)ws_size;

    const float* x      = (const float*)d_in[0];
    const float* g1     = (const float*)d_in[3];
    const float* b1     = (const float*)d_in[4];
    const float* w_qkv  = (const float*)d_in[5];
    const float* b_qkv  = (const float*)d_in[6];
    const float* w_proj = (const float*)d_in[7];
    const float* b_proj = (const float*)d_in[8];
    const float* g2     = (const float*)d_in[9];
    const float* b2     = (const float*)d_in[10];
    const float* w_fc1  = (const float*)d_in[11];
    const float* b_fc1  = (const float*)d_in[12];
    const float* w_fc2  = (const float*)d_in[13];
    const float* b_fc2  = (const float*)d_in[14];
    float* out = (float*)d_out;

    // workspace layout (all offsets 2^17-aligned)
    char* ws = (char*)d_ws;
    _Float16* wq   = (_Float16*)(ws + 0);                    //  768*256 f16
    _Float16* wp   = (_Float16*)(ws + 393216);               //  256*256 f16
    _Float16* w1   = (_Float16*)(ws + 524288);               // 1024*256 f16
    _Float16* w2   = (_Float16*)(ws + 1048576);              //  256*1024 f16
    _Float16* bufA = (_Float16*)(ws + 1572864);              // N_TOK*1024 f16
    _Float16* bufB = (_Float16*)(ws + 1572864 + (size_t)N_TOK * HID * 2); // N_TOK*256

    // 1) weights -> f16
    cvt_f32_f16_kernel<<<768, 256, 0, stream>>>(w_qkv, wq, NQKV * CCH);
    cvt_f32_f16_kernel<<<256, 256, 0, stream>>>(w_proj, wp, CCH * CCH);
    cvt_f32_f16_kernel<<<1024, 256, 0, stream>>>(w_fc1, w1, HID * CCH);
    cvt_f32_f16_kernel<<<1024, 256, 0, stream>>>(w_fc2, w2, CCH * HID);

    // 2) LN1 -> bufB (f16)
    ln_kernel<<<N_TOK, 256, 0, stream>>>(x, g1, b1, bufB);

    // 3) qkv = LN1 @ Wqkv^T + b  -> bufA (f16, ld=768)
    {
        int waves = (N_TOK / 32) * (NQKV / 32);
        gemm_wmma_kernel<0><<<waves / 8, 256, 0, stream>>>(
            bufB, wq, b_qkv, nullptr, bufA, NQKV, CCH);
    }

    // 4) windowed attention -> bufB (f16, ld=256)
    win_attn_kernel<<<2048 * HEADS, 32, 0, stream>>>(bufA, bufB);

    // 5) x1 = x + attn @ Wproj^T + b -> d_out (f32)
    {
        int waves = (N_TOK / 32) * (CCH / 32);
        gemm_wmma_kernel<1><<<waves / 8, 256, 0, stream>>>(
            bufB, wp, b_proj, x, out, CCH, CCH);
    }

    // 6) LN2(x1) -> bufB (f16)
    ln_kernel<<<N_TOK, 256, 0, stream>>>(out, g2, b2, bufB);

    // 7) hidden = gelu(LN2 @ Wfc1^T + b) -> bufA (f16, ld=1024)
    {
        int waves = (N_TOK / 32) * (HID / 32);
        gemm_wmma_kernel<2><<<waves / 8, 256, 0, stream>>>(
            bufB, w1, b_fc1, nullptr, bufA, HID, CCH);
    }

    // 8) d_out += hidden @ Wfc2^T + b
    {
        int waves = (N_TOK / 32) * (CCH / 32);
        gemm_wmma_kernel<3><<<waves / 8, 256, 0, stream>>>(
            bufA, w2, b_fc2, nullptr, out, CCH, HID);
    }
}
